// Enc_block_86071144612520
// MI455X (gfx1250) — matrix operand order repaired and checked by reading, not compiled, not run on hardware
//
#include <hip/hip_runtime.h>
#include <stdint.h>

// ---------------------------------------------------------------------------
// Types for CDNA5 WMMA (wave32, 16x16x32 bf16 -> f32)
// ---------------------------------------------------------------------------
typedef __bf16 bf16;
typedef __attribute__((ext_vector_type(16))) bf16  v16bf;
typedef __attribute__((ext_vector_type(8)))  float v8f;

#define CC       128          // Cout
#define EPSG     1e-20f
#define INV_TEMP 2.0f         // 1 / 0.5
#define HBINS    32768        // >= 21^3 voxel ids
#define FRAG     (32 * 16)    // bf16 elements per (kChunk, colTile) fragment block

// A-fragment element->K map (16-bit A 16x32, ISA 7.12.2):
// lanes 0-15: half=0, lanes 16-31: half=1
__device__ __forceinline__ int kmapA(int i, int half) {
  return ((i >> 3) << 4) + (((i >> 1) & 3) << 1) + (half << 3) + (i & 1);
}

// splitmix64-style hash -> uniform [0,1)
__device__ __forceinline__ float u01(uint32_t i, uint32_t j, uint32_t salt) {
  uint64_t z = (((uint64_t)i << 32) | (uint64_t)j) + (uint64_t)salt * 0x9E3779B97F4A7C15ull;
  z ^= z >> 30; z *= 0xBF58476D1CE4E5B9ull;
  z ^= z >> 27; z *= 0x94D049BB133111EBull;
  z ^= z >> 31;
  return (float)(uint32_t)(z >> 40) * (1.0f / 16777216.0f);
}

__device__ __forceinline__ float gumbelf(uint32_t i, uint32_t j) {
  float u = u01(i, j, 2u);
  return -logf(-logf(u + EPSG) + EPSG);
}

// ---------------------------------------------------------------------------
// Repack an f32 weight matrix W[Kd x Nd] into bf16 B-fragment order:
// Wb[(kb*Ntiles + ct)*32 + lane][16], element i holds
// W[(kb*32 + ((i>>3)<<4) + (half<<3) + (i&7)) * Nd + ct*16 + (lane&15)], zero-padded past Nd.
// One-time cost; makes the GEMM B-load a single aligned 32B vector load.
// ---------------------------------------------------------------------------
__global__ void repack_W(const float* __restrict__ W, bf16* __restrict__ Wb,
                         int Kd, int Nd, int Ntiles)
{
  const int idx = blockIdx.x * blockDim.x + threadIdx.x;
  const int total = (Kd / 32) * Ntiles * 32;
  if (idx >= total) return;
  const int lane = idx & 31;
  const int ct   = (idx >> 5) % Ntiles;
  const int kb   = (idx >> 5) / Ntiles;
  const int half = lane >> 4, l15 = lane & 15;
  const int c = ct * 16 + l15;
  bf16* dst = Wb + (size_t)idx * 16;
#pragma unroll
  for (int i = 0; i < 16; ++i) {
    int k = kb * 32 + ((i >> 3) << 4) + (half << 3) + (i & 7);
    dst[i] = (c < Nd) ? (bf16)W[(size_t)k * Nd + c] : (bf16)0.0f;
  }
}

// ---------------------------------------------------------------------------
// Generic WMMA GEMM: Y[M,Nd] = act(A[M,Kd] @ W + bias), W repacked bf16.
// One wave computes a 16 x (UT*16) output slab: the A fragment is loaded once
// per K-chunk and reused by UT back-to-back v_wmma ops. UT is a compile-time
// constant so the inner loop has no branches / accumulator shuffles.
// ---------------------------------------------------------------------------
template <int UT>
__global__ __launch_bounds__(32) void gemm_wmma(
    const float* __restrict__ A, const bf16* __restrict__ Wb,
    const float* __restrict__ bias, float* __restrict__ Y,
    int M, int Kd, int Nd, int Ntiles, int relu)
{
  const int lane = threadIdx.x;
  const int half = lane >> 4;
  const int l15  = lane & 15;
  const int ct0  = blockIdx.x * UT;
  const int row0 = blockIdx.y * 16;

  v8f acc[UT] = {};
  for (int kc = 0; kc < Kd; kc += 32) {
    const float* Arow = A + (size_t)(row0 + l15) * Kd + kc;
    __builtin_prefetch(Arow + 32, 0, 1);           // global_prefetch_b8
    v16bf a;
#pragma unroll
    for (int i = 0; i < 16; ++i) a[i] = (bf16)Arow[kmapA(i, half)];
    const bf16* Bk = Wb + ((size_t)(kc >> 5) * Ntiles + ct0) * FRAG + (size_t)lane * 16;
#pragma unroll
    for (int u = 0; u < UT; ++u) {
      v16bf b = *(const v16bf*)(Bk + (size_t)u * FRAG);
      acc[u] = __builtin_amdgcn_wmma_f32_16x16x32_bf16(false, a, false, b,
                                                       (short)0, acc[u], false, false);
    }
  }
#pragma unroll
  for (int u = 0; u < UT; ++u) {
    const int c = (ct0 + u) * 16 + l15;
    if (c < Nd) {
      float bv = bias ? bias[c] : 0.0f;
#pragma unroll
      for (int j = 0; j < 8; ++j) {
        int m = row0 + j + (half << 3);            // C/D layout: row = vgpr + 8*half
        float v = acc[u][j] + bv;
        if (relu) v = fmaxf(v, 0.0f);
        Y[(size_t)m * Nd + c] = v;
      }
    }
  }
}

// ---------------------------------------------------------------------------
// emb += uniform*0.001 (hash RNG stand-in for jax PRNG)
// ---------------------------------------------------------------------------
__global__ void k_embnoise(float* __restrict__ emb, int n) {
  int t = blockIdx.x * blockDim.x + threadIdx.x;
  if (t < n) emb[t] += u01((uint32_t)(t / 10), (uint32_t)(t % 10), 1u) * 0.001f;
}

// ---------------------------------------------------------------------------
// KNN top-16 (smallest squared distance) — one block per query point.
// per-thread register top-k over a strided subset + LDS tree merge.
// ---------------------------------------------------------------------------
__global__ __launch_bounds__(256) void knn_topk(
    const float* __restrict__ pos, int* __restrict__ srcIdx,
    float* __restrict__ wEdge, int N)
{
  __shared__ float sP[256 * 3];
  __shared__ float sVal[256 * 16];
  __shared__ int   sIdx[256 * 16];
  const int n = blockIdx.x, t = threadIdx.x;
  const float qx = pos[n * 3 + 0], qy = pos[n * 3 + 1], qz = pos[n * 3 + 2];

  float v[16]; int id[16];                 // sorted descending, v[0] = worst kept
#pragma unroll
  for (int q = 0; q < 16; ++q) { v[q] = INFINITY; id[q] = -1; }

  for (int jt = 0; jt < N; jt += 256) {
    for (int r = t; r < 256 * 3; r += 256) sP[r] = pos[jt * 3 + r];
    __syncthreads();
    int j = jt + t;
    if (j != n) {                          // diag excluded (set to inf in ref)
      float dx = qx - sP[t * 3 + 0], dy = qy - sP[t * 3 + 1], dz = qz - sP[t * 3 + 2];
      float d = dx * dx + dy * dy + dz * dz;
      if (d < v[0]) {
        v[0] = d; id[0] = j;
#pragma unroll
        for (int q = 0; q < 15; ++q)
          if (v[q] < v[q + 1]) { float tv = v[q]; v[q] = v[q + 1]; v[q + 1] = tv;
                                 int ti = id[q]; id[q] = id[q + 1]; id[q + 1] = ti; }
      }
    }
    __syncthreads();
  }
  // publish ascending
#pragma unroll
  for (int q = 0; q < 16; ++q) { sVal[t * 16 + q] = v[15 - q]; sIdx[t * 16 + q] = id[15 - q]; }
  __syncthreads();
  for (int stride = 128; stride >= 1; stride >>= 1) {
    if (t < stride) {
      const int ba = t * 16, bb = (t + stride) * 16;
      float mv[16]; int mi[16];
      int pa = 0, pb = 0;
#pragma unroll
      for (int q = 0; q < 16; ++q) {
        float av = sVal[ba + pa], bv = sVal[bb + pb];
        if (av <= bv) { mv[q] = av; mi[q] = sIdx[ba + pa]; ++pa; }
        else          { mv[q] = bv; mi[q] = sIdx[bb + pb]; ++pb; }
      }
#pragma unroll
      for (int q = 0; q < 16; ++q) { sVal[ba + q] = mv[q]; sIdx[ba + q] = mi[q]; }
    }
    __syncthreads();
  }
  if (t < 16) {
    srcIdx[n * 32 + 16 + t] = sIdx[t];     // knn edges occupy slots 16..31
    wEdge [n * 32 + 16 + t] = 1.0f;
  }
}

// ---------------------------------------------------------------------------
// Pass A of gumbel softmax (axis 0): per-column online logsumexp.
// noisy(i,j) never materialized; recomputed from emb + hash gumbel.
// ---------------------------------------------------------------------------
__global__ __launch_bounds__(256) void colstats(
    const float* __restrict__ emb, float* __restrict__ colmax,
    float* __restrict__ colsum, int N)
{
  __shared__ float sE[256 * 10];
  const int j = blockIdx.x * 256 + threadIdx.x;
  float ej[10];
#pragma unroll
  for (int d = 0; d < 10; ++d) ej[d] = emb[(size_t)j * 10 + d];

  float m = -INFINITY, s = 0.0f;
  for (int it = 0; it < N; it += 256) {
    for (int r = threadIdx.x; r < 256 * 10; r += 256) sE[r] = emb[(size_t)it * 10 + r];
    __syncthreads();
    for (int ii = 0; ii < 256; ++ii) {
      int i = it + ii;
      float d2 = 0.0f;
#pragma unroll
      for (int d = 0; d < 10; ++d) { float df = sE[ii * 10 + d] - ej[d]; d2 += df * df; }
      float val = (logf(__expf(-d2) + EPSG) + gumbelf((uint32_t)i, (uint32_t)j)) * INV_TEMP;
      if (val > m) { s = s * __expf(m - val) + 1.0f; m = val; }
      else         { s += __expf(val - m); }
    }
    __syncthreads();
  }
  colmax[j] = m;
  colsum[j] = s;
}

// ---------------------------------------------------------------------------
// Pass B: per-row top-16 of probs[i,j] = exp(noisy/T - colmax[j]) / colsum[j]
// ---------------------------------------------------------------------------
__global__ __launch_bounds__(256) void soft_topk(
    const float* __restrict__ emb, const float* __restrict__ colmax,
    const float* __restrict__ colsum, int* __restrict__ srcIdx,
    float* __restrict__ wEdge, int N)
{
  __shared__ float sE[256 * 10];
  __shared__ float sM[256], sS[256];
  __shared__ float sVal[256 * 16];
  __shared__ int   sIdx[256 * 16];
  const int i = blockIdx.x, t = threadIdx.x;
  float ei[10];
#pragma unroll
  for (int d = 0; d < 10; ++d) ei[d] = emb[(size_t)i * 10 + d];

  float v[16]; int id[16];                 // sorted ascending, v[0] = worst kept
#pragma unroll
  for (int q = 0; q < 16; ++q) { v[q] = -INFINITY; id[q] = -1; }

  for (int jt = 0; jt < N; jt += 256) {
    for (int r = t; r < 256 * 10; r += 256) sE[r] = emb[(size_t)jt * 10 + r];
    sM[t] = colmax[jt + t];
    sS[t] = colsum[jt + t];
    __syncthreads();
    int j = jt + t;
    float d2 = 0.0f;
#pragma unroll
    for (int d = 0; d < 10; ++d) { float df = sE[t * 10 + d] - ei[d]; d2 += df * df; }
    float noisy = (logf(__expf(-d2) + EPSG) + gumbelf((uint32_t)i, (uint32_t)j)) * INV_TEMP;
    float p = __expf(noisy - sM[t]) / sS[t];
    if (p > v[0]) {
      v[0] = p; id[0] = j;
#pragma unroll
      for (int q = 0; q < 15; ++q)
        if (v[q] > v[q + 1]) { float tv = v[q]; v[q] = v[q + 1]; v[q + 1] = tv;
                               int ti = id[q]; id[q] = id[q + 1]; id[q + 1] = ti; }
    }
    __syncthreads();
  }
  // publish descending, merge keeping largest
#pragma unroll
  for (int q = 0; q < 16; ++q) { sVal[t * 16 + q] = v[15 - q]; sIdx[t * 16 + q] = id[15 - q]; }
  __syncthreads();
  for (int stride = 128; stride >= 1; stride >>= 1) {
    if (t < stride) {
      const int ba = t * 16, bb = (t + stride) * 16;
      float mv[16]; int mi[16];
      int pa = 0, pb = 0;
#pragma unroll
      for (int q = 0; q < 16; ++q) {
        float av = sVal[ba + pa], bv = sVal[bb + pb];
        if (av >= bv) { mv[q] = av; mi[q] = sIdx[ba + pa]; ++pa; }
        else          { mv[q] = bv; mi[q] = sIdx[bb + pb]; ++pb; }
      }
#pragma unroll
      for (int q = 0; q < 16; ++q) { sVal[ba + q] = mv[q]; sIdx[ba + q] = mi[q]; }
    }
    __syncthreads();
  }
  if (t < 16) {
    srcIdx[i * 32 + t] = sIdx[t];          // soft edges occupy slots 0..15
    wEdge [i * 32 + t] = sVal[t];
  }
}

// ---------------------------------------------------------------------------
// In-LDS 32x128 @ 128x128 GEMM helper for the fused attention kernel.
// 8 waves x 2 tiles each = 16 (2 row-tiles x 8 col-tiles). B comes from the
// repacked bf16 fragment buffer (Ntiles = 8): one 32B vector load per wmma.
// ---------------------------------------------------------------------------
__device__ __forceinline__ void gemm32x128(
    const bf16* __restrict__ Albs, const bf16* __restrict__ Wb,
    const float* __restrict__ bias, bool relu,
    float* __restrict__ outF, bf16* __restrict__ outB, int wave, int lane)
{
  const int half = lane >> 4, l15 = lane & 15;
#pragma unroll
  for (int tt = 0; tt < 2; ++tt) {
    const int tile = wave + tt * 8;
    const int rt = tile >> 3, ct = tile & 7;
    v8f acc = {};
#pragma unroll
    for (int kc = 0; kc < CC; kc += 32) {
      const bf16* Ar = Albs + (rt * 16 + l15) * CC + kc;
      v16bf a;
#pragma unroll
      for (int i = 0; i < 16; ++i) a[i] = Ar[kmapA(i, half)];
      v16bf b = *(const v16bf*)(Wb + ((size_t)(kc >> 5) * 8 + ct) * FRAG
                                   + (size_t)lane * 16);
      acc = __builtin_amdgcn_wmma_f32_16x16x32_bf16(false, a, false, b,
                                                    (short)0, acc, false, false);
    }
    const int c = ct * 16 + l15;
    const float bv = bias ? bias[c] : 0.0f;
#pragma unroll
    for (int j = 0; j < 8; ++j) {
      const int r = rt * 16 + j + (half << 3);
      float val = acc[j] + bv;
      if (relu) val = fmaxf(val, 0.0f);
      if (outF) outF[r * CC + c] = val;
      if (outB) outB[r * CC + c] = (bf16)val;
    }
  }
}

// ---------------------------------------------------------------------------
// Fused PointTransformerConv: each node has exactly 32 contiguous edges
// (16 soft + 16 knn). One workgroup (256 thr / 8 waves) handles 16 nodes,
// per node: pos_nn MLP, attn_nn MLP (3 WMMA GEMMs), per-channel softmax,
// weighted aggregation. Logits/delta live only in LDS.
// ---------------------------------------------------------------------------
__global__ __launch_bounds__(256) void attn_fused(
    const float* __restrict__ pos, const int* __restrict__ srcIdx,
    const float* __restrict__ wEdge,
    const float* __restrict__ Aps, const float* __restrict__ Apd,
    const float* __restrict__ Vv,
    const float* __restrict__ Wp1, const float* __restrict__ bp1,
    const bf16* __restrict__ Wp2b, const float* __restrict__ bp2,
    const bf16* __restrict__ Wa1b, const float* __restrict__ ba1,
    const bf16* __restrict__ Wa2b, const float* __restrict__ ba2,
    float* __restrict__ H, int N, int nodesPerBlock)
{
  __shared__ bf16  sQ[32 * CC];      // pos_nn hidden (bf16)
  __shared__ float sDelta[32 * CC];  // pos_nn output (f32, reused in message)
  __shared__ bf16  sT[32 * CC];      // a_dst - a_src + delta (bf16)
  __shared__ bf16  sH2[32 * CC];     // attn_nn hidden (bf16)
  __shared__ float sL[32 * CC];      // logits (f32)
  __shared__ int   sSrc[32];
  __shared__ float sW[32];
  __shared__ float sPd[32 * 3];

  const int tid = threadIdx.x;
  const int wave = tid >> 5, lane = tid & 31;

  for (int nb = 0; nb < nodesPerBlock; ++nb) {
    const int n = blockIdx.x * nodesPerBlock + nb;
    if (n >= N) break;

    if (tid < 32) {
      int s = srcIdx[n * 32 + tid];
      sSrc[tid] = s;
      sW[tid] = wEdge[n * 32 + tid];
#pragma unroll
      for (int d = 0; d < 3; ++d)
        sPd[tid * 3 + d] = pos[(size_t)n * 3 + d] - pos[(size_t)s * 3 + d];
    }
    __syncthreads();

    // stage 1: Q = relu(posdiff @ Wp1 + bp1)   (K=3, plain VALU FMAs)
    {
      const int e = tid >> 3, c0 = (tid & 7) * 16;
      const float p0 = sPd[e * 3 + 0], p1 = sPd[e * 3 + 1], p2 = sPd[e * 3 + 2];
#pragma unroll
      for (int c = c0; c < c0 + 16; ++c) {
        float q = bp1[c] + p0 * Wp1[0 * CC + c] + p1 * Wp1[1 * CC + c] + p2 * Wp1[2 * CC + c];
        sQ[e * CC + c] = (bf16)fmaxf(q, 0.0f);
      }
    }
    __syncthreads();

    // stage 2: Delta = Q @ Wp2 + bp2
    gemm32x128(sQ, Wp2b, bp2, false, sDelta, nullptr, wave, lane);
    __syncthreads();

    // stage 3: T = a_dst - a_src + Delta
    {
      const int e = tid >> 3, c0 = (tid & 7) * 16;
      const int s = sSrc[e];
#pragma unroll
      for (int c = c0; c < c0 + 16; ++c) {
        float tv = Apd[(size_t)n * CC + c] - Aps[(size_t)s * CC + c] + sDelta[e * CC + c];
        sT[e * CC + c] = (bf16)tv;
      }
    }
    __syncthreads();

    // stage 4: H2 = relu(T @ Wa1 + ba1)
    gemm32x128(sT, Wa1b, ba1, true, nullptr, sH2, wave, lane);
    __syncthreads();

    // stage 5: L = H2 @ Wa2 + ba2 (logits)
    gemm32x128(sH2, Wa2b, ba2, false, sL, nullptr, wave, lane);
    __syncthreads();

    // stage 6: per-channel softmax over the node's 32 edges + aggregation
    if (tid < CC) {
      const int c = tid;
      float m = -INFINITY;
#pragma unroll 1
      for (int e = 0; e < 32; ++e) m = fmaxf(m, sL[e * CC + c]);
      float ssum = 0.0f;
#pragma unroll 1
      for (int e = 0; e < 32; ++e) ssum += __expf(sL[e * CC + c] - m);
      const float inv = 1.0f / (ssum + 1e-16f);
      float acc = 0.0f;
#pragma unroll 1
      for (int e = 0; e < 32; ++e) {
        float attn = __expf(sL[e * CC + c] - m) * inv;
        int s = sSrc[e];
        acc += attn * (Vv[(size_t)s * CC + c] + sDelta[e * CC + c]) * sW[e];
      }
      H[(size_t)n * CC + c] = acc;
    }
    __syncthreads();
  }
}

// ---------------------------------------------------------------------------
// max_pool_neighbor_x: pooled = max over this node's 32 sources, then
// elementwise max with self.
// ---------------------------------------------------------------------------
__global__ __launch_bounds__(CC) void maxpool(
    const float* __restrict__ H2n, const int* __restrict__ srcIdx,
    float* __restrict__ Hfin, int N)
{
  const int n = blockIdx.x, c = threadIdx.x;
  float m = H2n[(size_t)n * CC + c];
  float pooled = -INFINITY;
#pragma unroll 1
  for (int e = 0; e < 32; ++e) {
    int s = srcIdx[n * 32 + e];
    pooled = fmaxf(pooled, H2n[(size_t)s * CC + c]);
  }
  Hfin[(size_t)n * CC + c] = fmaxf(m, pooled);
}

// ---------------------------------------------------------------------------
// GridSampling(0.5): histogram -> rank scan (== sorted-unique inverse) ->
// atomic scatter mean directly into d_out.
// ---------------------------------------------------------------------------
__global__ void k_init(float* __restrict__ out, int out_n,
                       int* __restrict__ hist, int hn,
                       unsigned* __restrict__ minb, int* __restrict__ nvmax)
{
  int t = blockIdx.x * blockDim.x + threadIdx.x;
  int stride = gridDim.x * blockDim.x;
  for (int k = t; k < out_n; k += stride) out[k] = 0.0f;
  for (int k = t; k < hn; k += stride) hist[k] = 0;
  if (t < 3) { minb[t] = 0x7F800000u; nvmax[t] = 0; }
}

__global__ void k_posmin(const float* __restrict__ pos, unsigned* __restrict__ minb, int N) {
  int t = blockIdx.x * blockDim.x + threadIdx.x;
  if (t < N * 3) atomicMin(&minb[t % 3], __float_as_uint(pos[t]));  // pos >= 0
}

__device__ __forceinline__ void voxel3(const float* pos, int p, const unsigned* minb,
                                       int* v0, int* v1, int* v2) {
  *v0 = (int)floorf((pos[(size_t)p * 3 + 0] - __uint_as_float(minb[0])) * 2.0f);
  *v1 = (int)floorf((pos[(size_t)p * 3 + 1] - __uint_as_float(minb[1])) * 2.0f);
  *v2 = (int)floorf((pos[(size_t)p * 3 + 2] - __uint_as_float(minb[2])) * 2.0f);
}

__global__ void k_vmax(const float* __restrict__ pos, const unsigned* __restrict__ minb,
                       int* __restrict__ nvmax, int N) {
  int p = blockIdx.x * blockDim.x + threadIdx.x;
  if (p < N) {
    int v0, v1, v2; voxel3(pos, p, minb, &v0, &v1, &v2);
    atomicMax(&nvmax[0], v0); atomicMax(&nvmax[1], v1); atomicMax(&nvmax[2], v2);
  }
}

__global__ void k_hist(const float* __restrict__ pos, const unsigned* __restrict__ minb,
                       const int* __restrict__ nvmax, int* __restrict__ hist, int N) {
  int p = blockIdx.x * blockDim.x + threadIdx.x;
  if (p < N) {
    int v0, v1, v2; voxel3(pos, p, minb, &v0, &v1, &v2);
    int vid = (v0 * (nvmax[1] + 1) + v1) * (nvmax[2] + 1) + v2;
    atomicAdd(&hist[vid], 1);
  }
}

__global__ __launch_bounds__(1024) void k_scan(const int* __restrict__ hist,
                                               int* __restrict__ rank) {
  __shared__ int part[1024];
  const int t = threadIdx.x;
  const int base = t * (HBINS / 1024);
  int cnt = 0;
#pragma unroll 1
  for (int q = 0; q < HBINS / 1024; ++q) cnt += (hist[base + q] > 0);
  part[t] = cnt;
  __syncthreads();
  for (int off = 1; off < 1024; off <<= 1) {
    int vme = part[t];
    int vpr = (t >= off) ? part[t - off] : 0;
    __syncthreads();
    part[t] = vme + vpr;
    __syncthreads();
  }
  int run = (t > 0) ? part[t - 1] : 0;     // exclusive prefix of flags
#pragma unroll 1
  for (int q = 0; q < HBINS / 1024; ++q) {
    rank[base + q] = run;                  // rank == index in sorted unique
    run += (hist[base + q] > 0);
  }
}

__global__ __launch_bounds__(CC) void k_scatter(
    const float* __restrict__ pos, const float* __restrict__ Hfin,
    const int* __restrict__ rank, const unsigned* __restrict__ minb,
    const int* __restrict__ nvmax, float* __restrict__ out, int N)
{
  const int p = blockIdx.x, c = threadIdx.x;
  int v0, v1, v2; voxel3(pos, p, minb, &v0, &v1, &v2);
  const int vid = (v0 * (nvmax[1] + 1) + v1) * (nvmax[2] + 1) + v2;
  const int r = rank[vid];
  float* xo = out;
  float* po = out + (size_t)N * CC;
  float* co = po + (size_t)N * 3;
  atomicAdd(&xo[(size_t)r * CC + c], Hfin[(size_t)p * CC + c]);
  if (c < 3) atomicAdd(&po[(size_t)r * 3 + c], pos[(size_t)p * 3 + c]);
  if (c == 0) atomicAdd(&co[r], 1.0f);
}

__global__ __launch_bounds__(CC) void k_final(float* __restrict__ out, int N) {
  const int r = blockIdx.x, c = threadIdx.x;
  float* xo = out;
  float* po = out + (size_t)N * CC;
  float* co = po + (size_t)N * 3;
  const float denom = fmaxf(co[r], 1.0f);
  xo[(size_t)r * CC + c] /= denom;
  if (c < 3) po[(size_t)r * 3 + c] /= denom;
}

// ---------------------------------------------------------------------------
// Host launcher
// ---------------------------------------------------------------------------
extern "C" void kernel_launch(void* const* d_in, const int* in_sizes, int n_in,
                              void* d_out, int out_size, void* d_ws, size_t ws_size,
                              hipStream_t stream)
{
  const float* x    = (const float*)d_in[0];
  const float* pos  = (const float*)d_in[1];
  const float* Wg1  = (const float*)d_in[2];
  const float* bg1  = (const float*)d_in[3];
  const float* Wg2  = (const float*)d_in[4];
  const float* bg2  = (const float*)d_in[5];
  const float* Wlin = (const float*)d_in[6];
  const float* Wsrc = (const float*)d_in[7];
  const float* Wdst = (const float*)d_in[8];
  const float* Wp1  = (const float*)d_in[9];
  const float* bp1  = (const float*)d_in[10];
  const float* Wp2  = (const float*)d_in[11];
  const float* bp2  = (const float*)d_in[12];
  const float* Wa1  = (const float*)d_in[13];
  const float* ba1  = (const float*)d_in[14];
  const float* Wa2  = (const float*)d_in[15];
  const float* ba2  = (const float*)d_in[16];
  const float* Wd   = (const float*)d_in[17];
  const float* bd   = (const float*)d_in[18];

  const int N   = in_sizes[1] / 3;     // 8192
  const int Cin = in_sizes[0] / N;     // 64

  // workspace carve-up (256B aligned)
  char* wp = (char*)d_ws;
  auto alloc = [&](size_t bytes) {
    char* p = wp;
    wp += (bytes + 255) & ~(size_t)255;
    return (void*)p;
  };
  float*    embh   = (float*)alloc((size_t)N * Cin * 4);
  float*    emb    = (float*)alloc((size_t)N * 10 * 4);
  float*    Aps    = (float*)alloc((size_t)N * CC * 4);
  float*    Apd    = (float*)alloc((size_t)N * CC * 4);
  float*    Vv     = (float*)alloc((size_t)N * CC * 4);
  float*    H      = (float*)alloc((size_t)N * CC * 4);
  float*    H2n    = (float*)alloc((size_t)N * CC * 4);
  float*    Hfin   = (float*)alloc((size_t)N * CC * 4);
  float*    colmax = (float*)alloc((size_t)N * 4);
  float*    colsum = (float*)alloc((size_t)N * 4);
  float*    wE     = (float*)alloc((size_t)N * 32 * 4);
  int*      srcI   = (int*)alloc((size_t)N * 32 * 4);
  int*      hist   = (int*)alloc((size_t)HBINS * 4);
  int*      rank   = (int*)alloc((size_t)HBINS * 4);
  unsigned* minb   = (unsigned*)alloc(256);
  int*      nvmax  = (int*)alloc(256);
  // repacked bf16 weight fragment buffers
  const int tiCin = Cin / 16;          // 4  (64 cols)
  const int tiCC  = CC / 16;           // 8  (128 cols)
  bf16* Wg1b  = (bf16*)alloc((size_t)(Cin / 32) * tiCin * FRAG * 2);
  bf16* Wg2b  = (bf16*)alloc((size_t)(Cin / 32) * 1     * FRAG * 2);
  bf16* Wsrcb = (bf16*)alloc((size_t)(Cin / 32) * tiCC  * FRAG * 2);
  bf16* Wdstb = (bf16*)alloc((size_t)(Cin / 32) * tiCC  * FRAG * 2);
  bf16* Wlinb = (bf16*)alloc((size_t)(Cin / 32) * tiCC  * FRAG * 2);
  bf16* Wp2b  = (bf16*)alloc((size_t)(CC / 32) * tiCC * FRAG * 2);
  bf16* Wa1b  = (bf16*)alloc((size_t)(CC / 32) * tiCC * FRAG * 2);
  bf16* Wa2b  = (bf16*)alloc((size_t)(CC / 32) * tiCC * FRAG * 2);
  bf16* Wdb   = (bf16*)alloc((size_t)(CC / 32) * tiCC * FRAG * 2);

  // 0) zero outputs / control buffers (graph-replay safe)
  k_init<<<256, 256, 0, stream>>>((float*)d_out, out_size, hist, HBINS, minb, nvmax);

  // 0b) repack weights into bf16 B-fragment order
  auto rp = [&](const float* W, bf16* Wb, int Kd, int Nd, int Nt) {
    int total = (Kd / 32) * Nt * 32;
    repack_W<<<(total + 255) / 256, 256, 0, stream>>>(W, Wb, Kd, Nd, Nt);
  };
  rp(Wg1, Wg1b, Cin, Cin, tiCin);
  rp(Wg2, Wg2b, Cin, 10, 1);
  rp(Wsrc, Wsrcb, Cin, CC, tiCC);
  rp(Wdst, Wdstb, Cin, CC, tiCC);
  rp(Wlin, Wlinb, Cin, CC, tiCC);
  rp(Wp2, Wp2b, CC, CC, tiCC);
  rp(Wa1, Wa1b, CC, CC, tiCC);
  rp(Wa2, Wa2b, CC, CC, tiCC);
  rp(Wd, Wdb, CC, CC, tiCC);

  // 1) emb MLP: relu(x@Wg1+bg1) @ Wg2 + bg2, + hash noise
  gemm_wmma<4><<<dim3(tiCin / 4, N / 16), 32, 0, stream>>>(x, Wg1b, bg1, embh, N, Cin, Cin, tiCin, 1);
  gemm_wmma<1><<<dim3(1, N / 16), 32, 0, stream>>>(embh, Wg2b, bg2, emb, N, Cin, 10, 1, 0);
  k_embnoise<<<(N * 10 + 255) / 256, 256, 0, stream>>>(emb, N * 10);

  // 2) node projections x@{Wsrc,Wdst,Wlin}
  gemm_wmma<4><<<dim3(tiCC / 4, N / 16), 32, 0, stream>>>(x, Wsrcb, nullptr, Aps, N, Cin, CC, tiCC, 0);
  gemm_wmma<4><<<dim3(tiCC / 4, N / 16), 32, 0, stream>>>(x, Wdstb, nullptr, Apd, N, Cin, CC, tiCC, 0);
  gemm_wmma<4><<<dim3(tiCC / 4, N / 16), 32, 0, stream>>>(x, Wlinb, nullptr, Vv,  N, Cin, CC, tiCC, 0);

  // 3) KNN edges (slots 16..31) + gumbel soft edges (slots 0..15)
  knn_topk<<<N, 256, 0, stream>>>(pos, srcI, wE, N);
  colstats<<<N / 256, 256, 0, stream>>>(emb, colmax, colsum, N);
  soft_topk<<<N, 256, 0, stream>>>(emb, colmax, colsum, srcI, wE, N);

  // 4) fused PointTransformerConv
  attn_fused<<<N / 16, 256, 0, stream>>>(pos, srcI, wE, Aps, Apd, Vv,
                                         Wp1, bp1, Wp2b, bp2, Wa1b, ba1, Wa2b, ba2,
                                         H, N, 16);

  // 5) down: relu(H@Wd+bd), then neighbor max-pool
  gemm_wmma<4><<<dim3(tiCC / 4, N / 16), 32, 0, stream>>>(H, Wdb, bd, H2n, N, CC, CC, tiCC, 1);
  maxpool<<<N, CC, 0, stream>>>(H2n, srcI, Hfin, N);

  // 6) GridSampling(0.5) -> d_out = [x_out | pos_out | cnt]
  k_posmin<<<(N * 3 + 255) / 256, 256, 0, stream>>>(pos, minb, N);
  k_vmax<<<(N + 255) / 256, 256, 0, stream>>>(pos, minb, nvmax, N);
  k_hist<<<(N + 255) / 256, 256, 0, stream>>>(pos, minb, nvmax, hist, N);
  k_scan<<<1, 1024, 0, stream>>>(hist, rank);
  k_scatter<<<N, CC, 0, stream>>>(pos, Hfin, rank, minb, nvmax, (float*)d_out, N);
  k_final<<<N, CC, 0, stream>>>((float*)d_out, N);
}
